// NN_11166914969621
// MI455X (gfx1250) — matrix operand order repaired and checked by reading, not compiled, not run on hardware
//
#include <hip/hip_runtime.h>
#include <hip/hip_bf16.h>

#define NFEAT 128
#define NHID  256
#define NCLASS 16

typedef __attribute__((ext_vector_type(8)))  __bf16 v8bf;
typedef __attribute__((ext_vector_type(16))) __bf16 v16bf;
typedef __attribute__((ext_vector_type(8)))  float  v8f;

__device__ __forceinline__ unsigned short f2bf(float f) {
    union { float f; unsigned int u; } cv; cv.f = f;
    unsigned int u = cv.u;
    unsigned int r = u + 0x7FFFu + ((u >> 16) & 1u);   // round-to-nearest-even
    return (unsigned short)(r >> 16);
}

__device__ __forceinline__ float bf2f(unsigned int bits16) {
    union { unsigned int u; float f; } cv;
    cv.u = bits16 << 16;
    return cv.f;
}

// ---------------------------------------------------------------------------
// Cbf[M,NC](bf16) = A[M,K](bf16,row-major) x B[K,NC], Bt = B^T stored [NC,K] bf16.
// Tile: 128 rows x 64 cols per workgroup (256 thr = 8 waves),
// each wave: 16 rows x 64 cols = 4x v_wmma_f32_16x16x32_bf16 per K-step.
// Output written as bf16 so the SpMM gather working set fits in the 192MB L2.
// ---------------------------------------------------------------------------
__global__ __launch_bounds__(256) void gcn_gemm_bf16_wmma(
    const unsigned short* __restrict__ A,
    const unsigned short* __restrict__ Bt,
    unsigned short* __restrict__ Cbf,
    int M, int K, int NC)
{
    // +8 ushort (16B) pad per 32-elem row: 80B stride spreads LDS banks
    __shared__ __align__(16) unsigned short As[128 * 40];
    __shared__ __align__(16) unsigned short Bs[64 * 40];

    const int tid  = threadIdx.x;
    const int lane = tid & 31;
    const int wave = tid >> 5;
    const int row0 = blockIdx.x * 128;
    const int n0   = blockIdx.y * 64;
    const int hl   = (lane < 16) ? 0 : 1;   // half-wave select
    const int l16  = lane & 15;

    v8f acc[4];
    #pragma unroll
    for (int j = 0; j < 4; ++j)
        #pragma unroll
        for (int r = 0; r < 8; ++r) acc[j][r] = 0.0f;

    for (int k0 = 0; k0 < K; k0 += 32) {
        // ---- A tile 128x32 bf16: 2 x 16B per thread
        #pragma unroll
        for (int i = 0; i < 2; ++i) {
            int idx = tid * 2 + i;
            int rl  = idx >> 2;        // local row (32 bf16 = 4 uint4 per row)
            int cp  = idx & 3;
            uint4 v = make_uint4(0u, 0u, 0u, 0u);
            int gr = row0 + rl;
            if (gr < M) {
                const unsigned short* gp = A + (size_t)gr * K + k0 + cp * 8;
                v = *(const uint4*)gp;
                if (k0 + 32 < K)
                    __builtin_prefetch(gp + 32, 0, 3);   // global_prefetch_b8
            }
            *(uint4*)(&As[rl * 40 + cp * 8]) = v;
        }
        // ---- B tile (64 cols) x 32 K, from transposed weights: 1 x 16B per thread
        {
            int nl = tid >> 2;
            int cp = tid & 3;
            const unsigned short* gp = Bt + (size_t)(n0 + nl) * K + k0 + cp * 8;
            uint4 v = *(const uint4*)gp;
            if (k0 + 32 < K)
                __builtin_prefetch(gp + 32, 0, 3);
            *(uint4*)(&Bs[nl * 40 + cp * 8]) = v;
        }
        __syncthreads();

        // A fragment (16x32, 16-bit): lanes 0-15 row m hold K 0-15,
        // lanes 16-31 hold K 16-31.
        int arow = wave * 16 + l16;
        int alo  = hl ? 16 : 0;
        v8bf a0 = *(const v8bf*)(&As[arow * 40 + alo]);
        v8bf a1 = *(const v8bf*)(&As[arow * 40 + alo + 8]);
        v16bf afrag = __builtin_shufflevector(a0, a1,
            0,1,2,3,4,5,6,7,8,9,10,11,12,13,14,15);

        #pragma unroll
        for (int j = 0; j < 4; ++j) {
            // B fragment (32x16): lane n (0-15) holds K 0-15, lane n+16 holds K 16-31,
            // contiguous in the transposed-weight layout.
            int bcol = j * 16 + l16;
            int boff = hl ? 16 : 0;
            v8bf b0 = *(const v8bf*)(&Bs[bcol * 40 + boff]);
            v8bf b1 = *(const v8bf*)(&Bs[bcol * 40 + boff + 8]);
            v16bf bfrag = __builtin_shufflevector(b0, b1,
                0,1,2,3,4,5,6,7,8,9,10,11,12,13,14,15);

            // reuse_a hint: A operand identical to the previous (identical)
            // WMMA instruction for j >= 1 (ISA 7.12 VOP3P RA reuse rule)
            if (j == 0)
                acc[j] = __builtin_amdgcn_wmma_f32_16x16x32_bf16(
                    false, afrag, false, bfrag, (short)0, acc[j], false, false);
            else
                acc[j] = __builtin_amdgcn_wmma_f32_16x16x32_bf16(
                    false, afrag, false, bfrag, (short)0, acc[j], true, false);
        }
        __syncthreads();
    }

    // C/D layout: VGPR r -> M = r (lanes 0-15) or r+8 (lanes 16-31); N = lane%16
    #pragma unroll
    for (int j = 0; j < 4; ++j) {
        int ncol = n0 + j * 16 + l16;
        #pragma unroll
        for (int r = 0; r < 8; ++r) {
            int grow = row0 + wave * 16 + r + hl * 8;
            if (grow < M)
                Cbf[(size_t)grow * NC + ncol] = f2bf(acc[j][r]);
        }
    }
}

// ---------------------------------------------------------------------------
// Elementwise / prep kernels
// ---------------------------------------------------------------------------
__global__ void gcn_f32_to_bf16(const float* __restrict__ in,
                                unsigned short* __restrict__ out, int n) {
    int i = blockIdx.x * blockDim.x + threadIdx.x;
    if (i < n) out[i] = f2bf(in[i]);
}

__global__ void gcn_relu_to_bf16(const float* __restrict__ in,
                                 unsigned short* __restrict__ out, int n) {
    int i = blockIdx.x * blockDim.x + threadIdx.x;
    if (i < n) out[i] = f2bf(fmaxf(in[i], 0.0f));
}

// W[fin, fout] fp32 -> Wt[fout, fin] bf16
__global__ void gcn_transpose_w(const float* __restrict__ W,
                                unsigned short* __restrict__ Wt,
                                int fin, int fout) {
    int idx = blockIdx.x * blockDim.x + threadIdx.x;
    if (idx >= fin * fout) return;
    int i = idx / fout;   // input feature
    int o = idx % fout;   // output feature
    Wt[(size_t)o * fin + i] = f2bf(W[(size_t)i * fout + o]);
}

// accum[r, c] = bias[c]
__global__ void gcn_bias_init(float* __restrict__ out,
                              const float* __restrict__ bias, int total) {
    int i = blockIdx.x * blockDim.x + threadIdx.x;
    if (i < total) out[i] = bias[i % NHID];
}

// ---------------------------------------------------------------------------
// SpMM scatter: one wave per edge; lane covers 8 of 256 features.
// Gather source is bf16 (halves gather bytes; src+dst fit in 192MB L2),
// accumulation is fp32 via global_atomic_add_f32.
// ---------------------------------------------------------------------------
__global__ __launch_bounds__(256) void gcn_spmm_scatter(
    const int* __restrict__ rows, const int* __restrict__ cols,
    const float* __restrict__ vals, const unsigned short* __restrict__ src,
    float* __restrict__ dst, int E)
{
    int e = blockIdx.x * 8 + (threadIdx.x >> 5);
    if (e >= E) return;
    int lane = threadIdx.x & 31;
    int r = rows[e];
    int c = cols[e];
    float v = vals[e];
    // 8 bf16 features per lane = one 16B load for the whole row per wave
    uint4 p = *(const uint4*)(src + (size_t)c * NHID + lane * 8);
    float* d = dst + (size_t)r * NHID + lane * 8;
    atomicAdd(d + 0, bf2f(p.x & 0xFFFFu) * v);
    atomicAdd(d + 1, bf2f(p.x >> 16)     * v);
    atomicAdd(d + 2, bf2f(p.y & 0xFFFFu) * v);
    atomicAdd(d + 3, bf2f(p.y >> 16)     * v);
    atomicAdd(d + 4, bf2f(p.z & 0xFFFFu) * v);
    atomicAdd(d + 5, bf2f(p.z >> 16)     * v);
    atomicAdd(d + 6, bf2f(p.w & 0xFFFFu) * v);
    atomicAdd(d + 7, bf2f(p.w >> 16)     * v);
}

// ---------------------------------------------------------------------------
// Final: relu(h4) @ fc_w + fc_b, then log_softmax. fc_w staged in LDS (16 KB).
// ---------------------------------------------------------------------------
__global__ __launch_bounds__(256) void gcn_fc_logsoftmax(
    const float* __restrict__ h, const float* __restrict__ fcw,
    const float* __restrict__ fcb, float* __restrict__ out, int M)
{
    __shared__ float wfc[NHID * NCLASS];
    for (int i = threadIdx.x; i < NHID * NCLASS; i += 256)
        wfc[i] = fcw[i];
    __syncthreads();

    int row = blockIdx.x * 256 + threadIdx.x;
    if (row >= M) return;

    float acc[NCLASS];
    #pragma unroll
    for (int c = 0; c < NCLASS; ++c) acc[c] = fcb[c];

    const float* hr = h + (size_t)row * NHID;
    for (int k = 0; k < NHID; ++k) {
        float v = fmaxf(hr[k], 0.0f);
        #pragma unroll
        for (int c = 0; c < NCLASS; ++c)
            acc[c] = fmaf(v, wfc[k * NCLASS + c], acc[c]);
    }

    float m = acc[0];
    #pragma unroll
    for (int c = 1; c < NCLASS; ++c) m = fmaxf(m, acc[c]);
    float s = 0.0f;
    #pragma unroll
    for (int c = 0; c < NCLASS; ++c) s += __expf(acc[c] - m);
    float l = __logf(s);
    #pragma unroll
    for (int c = 0; c < NCLASS; ++c)
        out[(size_t)row * NCLASS + c] = acc[c] - m - l;
}

// ---------------------------------------------------------------------------
extern "C" void kernel_launch(void* const* d_in, const int* in_sizes, int n_in,
                              void* d_out, int out_size, void* d_ws, size_t ws_size,
                              hipStream_t stream) {
    const float* x       = (const float*)d_in[0];
    const float* adj_val = (const float*)d_in[1];
    const float* w[4] = {(const float*)d_in[2], (const float*)d_in[4],
                         (const float*)d_in[6], (const float*)d_in[8]};
    const float* b[4] = {(const float*)d_in[3], (const float*)d_in[5],
                         (const float*)d_in[7], (const float*)d_in[9]};
    const float* fcw     = (const float*)d_in[10];
    const float* fcb     = (const float*)d_in[11];
    const int* adj_row   = (const int*)d_in[12];
    const int* adj_col   = (const int*)d_in[13];

    const int Nn = in_sizes[0] / NFEAT;   // 100000
    const int E  = in_sizes[1];           // 3200000

    // ---- workspace layout
    char* ws = (char*)d_ws;
    size_t off = 0;
    auto take = [&](size_t bytes) -> char* {
        char* p = ws + off;
        off += (bytes + 255) & ~(size_t)255;
        return p;
    };
    unsigned short* xbf     = (unsigned short*)take((size_t)Nn * NFEAT * 2);
    unsigned short* hbf     = (unsigned short*)take((size_t)Nn * NHID * 2);
    unsigned short* support = (unsigned short*)take((size_t)Nn * NHID * 2); // bf16
    float* accum            = (float*)take((size_t)Nn * NHID * 4);
    unsigned short* wt[4];
    wt[0] = (unsigned short*)take((size_t)NFEAT * NHID * 2);
    wt[1] = (unsigned short*)take((size_t)NHID * NHID * 2);
    wt[2] = (unsigned short*)take((size_t)NHID * NHID * 2);
    wt[3] = (unsigned short*)take((size_t)NHID * NHID * 2);

    // ---- prep: bf16 input + transposed bf16 weights
    {
        int n = Nn * NFEAT;
        gcn_f32_to_bf16<<<(n + 255) / 256, 256, 0, stream>>>(x, xbf, n);
    }
    for (int l = 0; l < 4; ++l) {
        int fin = (l == 0) ? NFEAT : NHID;
        int n = fin * NHID;
        gcn_transpose_w<<<(n + 255) / 256, 256, 0, stream>>>(w[l], wt[l], fin, NHID);
    }

    // ---- 4 GCN layers
    const int gemmM = (Nn + 127) / 128;
    for (int l = 0; l < 4; ++l) {
        const unsigned short* inbf = (l == 0) ? xbf : hbf;
        const int Kdim = (l == 0) ? NFEAT : NHID;

        gcn_gemm_bf16_wmma<<<dim3(gemmM, NHID / 64), 256, 0, stream>>>(
            inbf, wt[l], support, Nn, Kdim, NHID);

        int tot = Nn * NHID;
        gcn_bias_init<<<(tot + 255) / 256, 256, 0, stream>>>(accum, b[l], tot);

        gcn_spmm_scatter<<<(E + 7) / 8, 256, 0, stream>>>(
            adj_row, adj_col, adj_val, support, accum, E);

        if (l < 3)
            gcn_relu_to_bf16<<<(tot + 255) / 256, 256, 0, stream>>>(accum, hbf, tot);
    }

    // ---- FC + log_softmax (ReLU applied inline)
    gcn_fc_logsoftmax<<<(Nn + 255) / 256, 256, 0, stream>>>(
        accum, fcw, fcb, (float*)d_out, Nn);
}
